// MultiHeadAttention_44461501448193
// MI455X (gfx1250) — matrix operand order repaired and checked by reading, not compiled, not run on hardware
//
#include <hip/hip_runtime.h>

#define BATCH 2
#define SEQ   2048
#define EMBD  1024
#define NH    16
#define HD    64

typedef __bf16 bf16;
typedef __attribute__((ext_vector_type(16))) __bf16 v16bf;
typedef __attribute__((ext_vector_type(8)))  __bf16 v8bf;
typedef __attribute__((ext_vector_type(8)))  float  v8f;
typedef __attribute__((ext_vector_type(4)))  unsigned int u32x4;
typedef __attribute__((ext_vector_type(8)))  int i32x8;
typedef __attribute__((ext_vector_type(4)))  int i32x4;

#ifndef __has_builtin
#define __has_builtin(x) 0
#endif
#if __has_builtin(__builtin_amdgcn_tensor_load_to_lds)
#define HAVE_TDM 1
#else
#define HAVE_TDM 0
#endif

// TDM config word0 of D# group1 for rows of 64 bf16 (128B = 32 DWORDs):
// data_size=1 (2B), pad_enable=1, pad_interval=4 (32 DWORDs), pad_amount=3
// (4 DWORDs = 8 bf16)  ->  LDS row stride 72 elements (144B, conflict-free).
#define TDMCFG 0x07110000u
#define LD72 72

__device__ __forceinline__ void tdm_load_2d(const bf16* gptr, unsigned lds_off,
                                            unsigned tile_w, unsigned tile_h,
                                            unsigned tensor_w, unsigned tensor_h,
                                            unsigned stride_elems) {
#if HAVE_TDM
  unsigned long long ga = (unsigned long long)(size_t)gptr;
  u32x4 g0 = {1u,  // count=1 (valid user descriptor)
              lds_off, (unsigned)(ga & 0xFFFFFFFFu),
              (unsigned)((ga >> 32) & 0x1FFFFFFu) | (2u << 30)};  // type=2
  i32x8 g1 = {(int)TDMCFG,
              (int)((tensor_w & 0xFFFFu) << 16),                  // dim0 lo
              (int)((tensor_w >> 16) | ((tensor_h & 0xFFFFu) << 16)),
              (int)((tensor_h >> 16) | (tile_w << 16)),           // tile_dim0
              (int)tile_h,                                        // tile_dim1
              (int)stride_elems,                                  // stride0 lo
              0, 0};
  i32x4 z4 = {0, 0, 0, 0};
#if defined(__clang_major__) && __clang_major__ >= 23
  i32x8 z8 = {0, 0, 0, 0, 0, 0, 0, 0};
  __builtin_amdgcn_tensor_load_to_lds(g0, g1, z4, z4, z8, 0);
#else
  __builtin_amdgcn_tensor_load_to_lds(g0, g1, z4, z4, 0);
#endif
#endif
}

__device__ __forceinline__ void tdm_wait() {
#if HAVE_TDM
  __builtin_amdgcn_s_wait_tensorcnt(0);
#endif
}

__device__ __forceinline__ v8f wmma_bf16(v16bf a, v16bf b, v8f c) {
  return __builtin_amdgcn_wmma_f32_16x16x32_bf16(false, a, false, b, (short)0, c,
                                                 false, false);
}

// A-fragment (16x32 bf16 tile, row-major in LDS, row stride LD elements).
template <int LD>
__device__ __forceinline__ v16bf load_frag_a(const bf16* base, int lane) {
  const bf16* p = base + (lane & 15) * LD + ((lane >> 4) << 3);
  v8bf lo = *(const v8bf*)p;
  v8bf hi = *(const v8bf*)(p + 16);
  return __builtin_shufflevector(lo, hi, 0, 1, 2, 3, 4, 5, 6, 7, 8, 9, 10, 11,
                                 12, 13, 14, 15);
}

// B-fragment from an [n][k] (transposed) LDS tile, row stride LD elements.
template <int LD>
__device__ __forceinline__ v16bf load_frag_b(const bf16* base, int lane) {
  const bf16* p = base + (lane & 15) * LD + ((lane >> 4) << 3);
  v8bf lo = *(const v8bf*)p;
  v8bf hi = *(const v8bf*)(p + 16);
  return __builtin_shufflevector(lo, hi, 0, 1, 2, 3, 4, 5, 6, 7, 8, 9, 10, 11,
                                 12, 13, 14, 15);
}

__global__ __launch_bounds__(256) void f32_to_bf16(const float* __restrict__ in,
                                                   bf16* __restrict__ out, int n) {
  int i = (blockIdx.x * 256 + threadIdx.x) * 4;
  if (i + 3 < n) {
    float4 x = *(const float4*)(in + i);
    out[i + 0] = (bf16)x.x;
    out[i + 1] = (bf16)x.y;
    out[i + 2] = (bf16)x.z;
    out[i + 3] = (bf16)x.w;
  } else {
    for (int j = i; j < n; ++j) out[j] = (bf16)in[j];
  }
}

// ---------------------------------------------------------------------------
// GEMM: C[M,N] = (A[M,K]*B[K,N] + bias[N]) * scale.  Block tile 128x64,
// K-step 64, double-buffered: TDM streams A tile i+1 into LDS while tile i is
// computed; B tile i+1 staged transposed (vectorized loads, unrolled scatter).
// 8 waves: 4(M) x 2(N), each owns a 32x32 accumulator. One barrier per K-step.
// ---------------------------------------------------------------------------
#define GBM 128
#define GBN 64
#define GBK 64

__global__ __launch_bounds__(256) void gemm_bf16_wmma(
    const bf16* __restrict__ A, const bf16* __restrict__ Bw,
    const float* __restrict__ bias, bf16* __restrict__ Cb,
    float* __restrict__ Cf, int Mdim, int Ndim, int Kdim, float scale) {
  __shared__ bf16 As[2][GBM][LD72];   // [m][k] padded rows (TDM-padded)
  __shared__ bf16 Bst[2][GBN][LD72];  // [n][k] transposed

  const int m0 = blockIdx.y * GBM;
  const int n0 = blockIdx.x * GBN;
  const int t = threadIdx.x;
  const int lane = t & 31;
  const int wave = t >> 5;
  const int wm = wave & 3;
  const int wn = wave >> 2;
  const int ntiles = Kdim / GBK;

  v8f acc[2][2] = {};

  // stage B tile (64k x 64n) transposed into Bst[buf]
  auto loadB = [&](int k0, int buf) {
#pragma unroll
    for (int j = 0; j < 2; ++j) {
      int chunk = t + j * 256;  // 0..511 : r = k-row, cb = 8-col block
      int r = chunk >> 3, cb = chunk & 7;
      v8bf x = *(const v8bf*)(Bw + (size_t)(k0 + r) * Ndim + n0 + cb * 8);
#pragma unroll
      for (int e = 0; e < 8; ++e) Bst[buf][cb * 8 + e][r] = x[e];
    }
  };
#if !HAVE_TDM
  auto loadA = [&](int k0, int buf) {
#pragma unroll
    for (int j = 0; j < 2; ++j) {
      int idx = t + j * 256;  // 0..511, 16 elems each
      int r = idx >> 2, sg = (idx & 3) * 16;
      const bf16* src = A + (size_t)(m0 + r) * Kdim + k0 + sg;
      *(v8bf*)&As[buf][r][sg] = *(const v8bf*)src;
      *(v8bf*)&As[buf][r][sg + 8] = *(const v8bf*)(src + 8);
    }
  };
#endif

  // prologue: fill buffer 0
#if HAVE_TDM
  if (wave == 0)
    tdm_load_2d(A + (size_t)m0 * Kdim, (unsigned)(size_t)&As[0][0][0], GBK, GBM,
                (unsigned)Kdim, (unsigned)Mdim, (unsigned)Kdim);
#else
  loadA(0, 0);
#endif
  loadB(0, 0);
  if (wave == 0) tdm_wait();
  __syncthreads();

  for (int i = 0; i < ntiles; ++i) {
    const int cur = i & 1, nxt = cur ^ 1;
    const int k1 = (i + 1) * GBK;
    if (i + 1 < ntiles) {
#if HAVE_TDM
      if (wave == 0)
        tdm_load_2d(A + (size_t)m0 * Kdim + k1, (unsigned)(size_t)&As[nxt][0][0],
                    GBK, GBM, (unsigned)Kdim, (unsigned)Mdim, (unsigned)Kdim);
#else
      loadA(k1, nxt);
#endif
      loadB(k1, nxt);
      if (i + 2 < ntiles)  // L2 hint for the B tile after next
        __builtin_prefetch(Bw + (size_t)(k1 + GBK + (t >> 3)) * Ndim + n0, 0, 1);
    }
    const int k0c = i * GBK; (void)k0c;
#pragma unroll
    for (int kk = 0; kk < GBK; kk += 32) {
      v16bf a0 = load_frag_a<LD72>(&As[cur][wm * 32][kk], lane);
      v16bf a1 = load_frag_a<LD72>(&As[cur][wm * 32 + 16][kk], lane);
      v16bf b0 = load_frag_b<LD72>(&Bst[cur][wn * 32][kk], lane);
      v16bf b1 = load_frag_b<LD72>(&Bst[cur][wn * 32 + 16][kk], lane);
      acc[0][0] = wmma_bf16(a0, b0, acc[0][0]);
      acc[0][1] = wmma_bf16(a0, b1, acc[0][1]);
      acc[1][0] = wmma_bf16(a1, b0, acc[1][0]);
      acc[1][1] = wmma_bf16(a1, b1, acc[1][1]);
    }
    if (wave == 0 && i + 1 < ntiles) tdm_wait();
    __syncthreads();
  }

#pragma unroll
  for (int mi = 0; mi < 2; ++mi)
#pragma unroll
    for (int ni = 0; ni < 2; ++ni)
#pragma unroll
      for (int j = 0; j < 8; ++j) {
        int row = m0 + wm * 32 + mi * 16 + j + ((lane >> 4) << 3);
        int col = n0 + wn * 32 + ni * 16 + (lane & 15);
        float v = (acc[mi][ni][j] + bias[col]) * scale;
        if (Cb) Cb[(size_t)row * Ndim + col] = (bf16)v;
        if (Cf) Cf[(size_t)row * Ndim + col] = v;
      }
}

// ---------------------------------------------------------------------------
// Flash attention, [B,S,E] bf16, E = h*64+d. 4 waves; wave w owns 16 q-rows.
// Q tile + K tiles streamed by TDM (K double-buffered against compute);
// V^T staged manually (TDM cannot transpose). One barrier per kv-step.
// ---------------------------------------------------------------------------
__global__ __launch_bounds__(128) void attn_wmma(const bf16* __restrict__ Q,
                                                 const bf16* __restrict__ K,
                                                 const bf16* __restrict__ V,
                                                 bf16* __restrict__ ctx) {
  __shared__ bf16 Qs[64][LD72];      // [q][d]
  __shared__ bf16 Ks[2][64][LD72];   // [kc][d]  (acts as B = K^T in [n][k])
  __shared__ bf16 Vst[2][64][LD72];  // [d][kc]  (V^T in [n][k])
  __shared__ bf16 Ps[64][LD72];      // per-wave strip of P, [q][kc]

  const int bh = blockIdx.y;
  const int b = bh >> 4, h = bh & (NH - 1);
  const int q0 = blockIdx.x * 64;
  const int t = threadIdx.x, lane = t & 31, wave = t >> 5;
  const size_t rs = EMBD;
  const int nt = SEQ / 64;

  const bf16* Qbase = Q + ((size_t)b * SEQ + q0) * EMBD + h * HD;
  const bf16* Kbase = K + (size_t)b * SEQ * EMBD + h * HD;
  const bf16* Vbase = V + (size_t)b * SEQ * EMBD + h * HD;

  auto loadVT = [&](int k0, int buf) {
#pragma unroll
    for (int j = 0; j < 4; ++j) {
      int chunk = t + j * 128;  // 0..511 : kc row, db = 8-col block of d
      int kc = chunk >> 3, db = chunk & 7;
      v8bf x = *(const v8bf*)(Vbase + (size_t)(k0 + kc) * rs + db * 8);
#pragma unroll
      for (int e = 0; e < 8; ++e) Vst[buf][db * 8 + e][kc] = x[e];
    }
  };
#if !HAVE_TDM
  auto loadRM = [&](const bf16* src, bf16(*dst)[LD72], int k0) {
#pragma unroll
    for (int j = 0; j < 2; ++j) {
      int i = t + j * 128;  // 0..255, 16 elems each
      int r = i >> 2, sg = (i & 3) * 16;
      const bf16* p = src + (size_t)(k0 + r) * rs + sg;
      *(v8bf*)&dst[r][sg] = *(const v8bf*)p;
      *(v8bf*)&dst[r][sg + 8] = *(const v8bf*)(p + 8);
    }
  };
#endif

  // prologue: Q tile + first K/V tiles
#if HAVE_TDM
  if (wave == 0) {
    tdm_load_2d(Qbase, (unsigned)(size_t)&Qs[0][0], 64, 64, EMBD, SEQ, EMBD);
    tdm_load_2d(Kbase, (unsigned)(size_t)&Ks[0][0][0], 64, 64, EMBD, SEQ, EMBD);
  }
#else
  loadRM(Qbase, &Qs[0], 0);
  loadRM(Kbase, &Ks[0][0], 0);
#endif
  loadVT(0, 0);
  if (wave == 0) tdm_wait();
  __syncthreads();

  float m_r[8], l_r[8];
#pragma unroll
  for (int j = 0; j < 8; ++j) { m_r[j] = -1e30f; l_r[j] = 0.f; }
  v8f c_acc[4] = {};

  for (int i = 0; i < nt; ++i) {
    const int cur = i & 1, nxt = cur ^ 1;
    const int k1 = (i + 1) * 64;
    if (i + 1 < nt) {
#if HAVE_TDM
      if (wave == 0)
        tdm_load_2d(Kbase + (size_t)k1 * rs, (unsigned)(size_t)&Ks[nxt][0][0],
                    64, 64, EMBD, SEQ, EMBD);
#else
      loadRM(Kbase, &Ks[nxt][0], k1);
#endif
      loadVT(k1, nxt);
    }

    // S = Q * K^T  (16 q-rows x 64 k-cols per wave); 1/sqrt(hd) folded into Q
    v8f s[4] = {};
#pragma unroll
    for (int kk = 0; kk < 64; kk += 32) {
      v16bf aq = load_frag_a<LD72>(&Qs[wave * 16][kk], lane);
#pragma unroll
      for (int ni = 0; ni < 4; ++ni) {
        v16bf bk = load_frag_b<LD72>(&Ks[cur][ni * 16][kk], lane);
        s[ni] = wmma_bf16(aq, bk, s[ni]);
      }
    }

    // online softmax; row = wave*16 + j + (lane>>4)*8, cols across 16 lanes
#pragma unroll
    for (int j = 0; j < 8; ++j) {
      float rmax = s[0][j];
#pragma unroll
      for (int ni = 1; ni < 4; ++ni) rmax = fmaxf(rmax, s[ni][j]);
#pragma unroll
      for (int off = 8; off >= 1; off >>= 1)
        rmax = fmaxf(rmax, __shfl_xor(rmax, off, 32));
      float mnew = fmaxf(m_r[j], rmax);
      float corr = __expf(m_r[j] - mnew);
      float psum = 0.f;
      int prow = wave * 16 + j + ((lane >> 4) << 3);
#pragma unroll
      for (int ni = 0; ni < 4; ++ni) {
        float pv = __expf(s[ni][j] - mnew);
        psum += pv;
        Ps[prow][ni * 16 + (lane & 15)] = (bf16)pv;
      }
#pragma unroll
      for (int off = 8; off >= 1; off >>= 1) psum += __shfl_xor(psum, off, 32);
      l_r[j] = l_r[j] * corr + psum;
      m_r[j] = mnew;
#pragma unroll
      for (int di = 0; di < 4; ++di) c_acc[di][j] *= corr;
    }

    // ctx += P * V   (P read back from this wave's own LDS strip)
#pragma unroll
    for (int kk = 0; kk < 64; kk += 32) {
      v16bf ap = load_frag_a<LD72>(&Ps[wave * 16][kk], lane);
#pragma unroll
      for (int di = 0; di < 4; ++di) {
        v16bf bv = load_frag_b<LD72>(&Vst[cur][di * 16][kk], lane);
        c_acc[di] = wmma_bf16(ap, bv, c_acc[di]);
      }
    }
    if (wave == 0 && i + 1 < nt) tdm_wait();
    __syncthreads();
  }

  bf16* out = ctx + ((size_t)b * SEQ + q0) * EMBD + h * HD;
#pragma unroll
  for (int di = 0; di < 4; ++di)
#pragma unroll
    for (int j = 0; j < 8; ++j) {
      int row = wave * 16 + j + ((lane >> 4) << 3);
      int col = di * 16 + (lane & 15);
      out[(size_t)row * rs + col] = (bf16)(c_acc[di][j] / l_r[j]);
    }
}

// ---------------------------------------------------------------------------
extern "C" void kernel_launch(void* const* d_in, const int* in_sizes, int n_in,
                              void* d_out, int out_size, void* d_ws,
                              size_t ws_size, hipStream_t stream) {
  (void)in_sizes; (void)n_in; (void)out_size; (void)ws_size;
  const float* q_f = (const float*)d_in[0];
  const float* k_f = (const float*)d_in[1];
  const float* v_f = (const float*)d_in[2];
  const float* Wq_f = (const float*)d_in[3];
  const float* bq = (const float*)d_in[4];
  const float* Wk_f = (const float*)d_in[5];
  const float* bk = (const float*)d_in[6];
  const float* Wv_f = (const float*)d_in[7];
  const float* bv = (const float*)d_in[8];
  const float* Wo_f = (const float*)d_in[9];
  const float* bo = (const float*)d_in[10];
  float* out = (float*)d_out;

  const size_t ME = (size_t)BATCH * SEQ * EMBD;  // 4M elems
  const size_t EE = (size_t)EMBD * EMBD;         // 1M elems
  bf16* p = (bf16*)d_ws;
  bf16 *qx = p, *kx = qx + ME, *vx = kx + ME;
  bf16 *wq = vx + ME, *wk = wq + EE, *wv = wk + EE, *wo = wv + EE;
  bf16 *Qp = wo + EE, *Kp = Qp + ME, *Vp = Kp + ME, *Cx = Vp + ME;

  auto cvt = [&](const float* src, bf16* dst, size_t n) {
    unsigned blocks = (unsigned)((n / 4 + 255) / 256);
    f32_to_bf16<<<dim3(blocks), dim3(256), 0, stream>>>(src, dst, (int)n);
  };
  cvt(q_f, qx, ME);
  cvt(k_f, kx, ME);
  cvt(v_f, vx, ME);
  cvt(Wq_f, wq, EE);
  cvt(Wk_f, wk, EE);
  cvt(Wv_f, wv, EE);
  cvt(Wo_f, wo, EE);

  dim3 gblk(256);
  dim3 ggrd(EMBD / GBN, (BATCH * SEQ) / GBM);  // (16, 32)
  const int M = BATCH * SEQ;
  gemm_bf16_wmma<<<ggrd, gblk, 0, stream>>>(qx, wq, bq, Qp, nullptr, M, EMBD,
                                            EMBD, 0.125f);  // 1/sqrt(HD)
  gemm_bf16_wmma<<<ggrd, gblk, 0, stream>>>(kx, wk, bk, Kp, nullptr, M, EMBD,
                                            EMBD, 1.0f);
  gemm_bf16_wmma<<<ggrd, gblk, 0, stream>>>(vx, wv, bv, Vp, nullptr, M, EMBD,
                                            EMBD, 1.0f);

  attn_wmma<<<dim3(SEQ / 64, BATCH * NH), dim3(128), 0, stream>>>(Qp, Kp, Vp,
                                                                  Cx);

  gemm_bf16_wmma<<<ggrd, gblk, 0, stream>>>(Cx, wo, bo, nullptr, out, M, EMBD,
                                            EMBD, 1.0f);
}
